// TripletHead4_22660247454414
// MI455X (gfx1250) — matrix operand order repaired and checked by reading, not compiled, not run on hardware
//
#include <hip/hip_runtime.h>
#include <hip/hip_bf16.h>

typedef __attribute__((ext_vector_type(16))) __bf16 v16bf;
typedef __attribute__((ext_vector_type(8)))  __bf16 v8bf;
typedef __attribute__((ext_vector_type(4)))  __bf16 v4bf;
typedef __attribute__((ext_vector_type(2)))  __bf16 v2bf;
typedef __attribute__((ext_vector_type(8)))  float  v8f;

#define BQ 16
#define PQ 4096
#define DQ 512
#define HQ 512
#define RQ 50
#define M_TOTAL (BQ*PQ)                 /* 65536 */
#define PRED_N ((size_t)M_TOTAL*RQ)     /* 3276800 */
#define M_IMG (PQ*RQ)                   /* 204800 per image */

#define MTILE 64                        /* rows per block */
#define HSTR  520                       /* hbuf row stride (bank spread) */
#define XSTR  40                        /* xtile row stride (bank spread) */
#define LDS_XTILE_B (MTILE * XSTR * 2)              /* 5120 B  */
#define LDS_HBUF_B  (MTILE * HSTR * 2)              /* 66560 B */
#define LDS_TOTAL_B (LDS_XTILE_B + LDS_HBUF_B)      /* 71680 B */

__device__ __forceinline__ __bf16 f2bf(float f) {
  unsigned u = __float_as_uint(f);
  unsigned r = (u + 0x7FFFu + ((u >> 16) & 1u)) >> 16;
  unsigned short s = (unsigned short)r;
  __bf16 b;
  __builtin_memcpy(&b, &s, 2);
  return b;
}

// Packed f32x2 -> bf16x2 (RTNE). Prefer the clang builtin; on gfx1250 where
// the builtin is not declared, emit v_cvt_pk_bf16_f32 directly (1 VALU op
// instead of ~8 for the manual round-to-nearest-even sequence).
__device__ __forceinline__ v2bf pk2(float a, float b) {
#if __has_builtin(__builtin_amdgcn_cvt_pk_bf16_f32)
  return __builtin_amdgcn_cvt_pk_bf16_f32(a, b);
#elif defined(__gfx1250__)
  unsigned o;
  asm("v_cvt_pk_bf16_f32 %0, %1, %2" : "=v"(o) : "v"(a), "v"(b));
  v2bf r;
  __builtin_memcpy(&r, &o, 4);
  return r;
#else
  v2bf r; r[0] = f2bf(a); r[1] = f2bf(b); return r;
#endif
}

union BF16x16 { v16bf v; v8bf h[2]; };
union BF16x8  { v8bf  v; v2bf p[4]; };

// ---------------------------------------------------------------------------
// Kernel 1: weight prep. w1 [1024,512] f32 -> w1t [512][1024] bf16 (n-major),
// w2 [512,50] f32 -> w2t [64][512] bf16 (n-major, zero-padded cols 50..63).
// Also zeroes the 11-float loss/metrics tail of d_out and 3 ws accumulators.
// ---------------------------------------------------------------------------
__global__ __launch_bounds__(256)
void th4_convert(const float* __restrict__ w1, const float* __restrict__ w2,
                 __bf16* __restrict__ w1t, __bf16* __restrict__ w2t,
                 float* __restrict__ wsacc, float* __restrict__ outtail) {
  const int i = blockIdx.x * 256 + threadIdx.x;
  const int N1 = 512 * 1024;
  const int N2 = 64 * 512;
  if (i < N1) {
    const int n = i >> 10, k = i & 1023;
    w1t[i] = f2bf(w1[k * 512 + n]);
  } else if (i < N1 + N2) {
    const int j = i - N1;
    const int n = j >> 9, k = j & 511;
    w2t[j] = f2bf(n < RQ ? w2[k * RQ + n] : 0.0f);
  } else if (i < N1 + N2 + 11) {
    outtail[i - (N1 + N2)] = 0.0f;
  } else if (i < N1 + N2 + 14) {
    wsacc[i - (N1 + N2 + 11)] = 0.0f;
  }
}

// ---------------------------------------------------------------------------
// Kernel 2: fused GEMM1(relu) + GEMM2(sigmoid) per 64-row tile, bf16 WMMA.
// Block = 256 threads (8 waves). Wave w owns H-columns [64w, 64w+64) for
// GEMM1 (4 M-subtiles x 4 N-subtiles = 16 accumulators); for GEMM2 each wave
// computes two 16x16 tiles of the 64x64 output.
// ---------------------------------------------------------------------------
__global__ __launch_bounds__(256)
void th4_gemm_fused(const float* __restrict__ pairf, const float* __restrict__ relf,
                    const __bf16* __restrict__ w1t, const float* __restrict__ b1,
                    const __bf16* __restrict__ w2t, const float* __restrict__ b2,
                    float* __restrict__ pred) {
  extern __shared__ char smem[];
  __bf16* xtile = (__bf16*)smem;                   // [64][XSTR]
  __bf16* hbuf  = (__bf16*)(smem + LDS_XTILE_B);   // [64][HSTR]

  const int tid  = threadIdx.x;
  const int w    = tid >> 5;
  const int lane = tid & 31;
  const int lnm  = lane & 15;
  const int hf   = lane >> 4;
  const size_t r0 = (size_t)blockIdx.x * MTILE;

  v8f acc[16] = {};   // [mi*4 + ni], mi in 0..3, ni in 0..3

  // Per-lane B-fragment row pointers into w1t for the 4 N-subtiles.
  const __bf16* bp[4];
#pragma unroll
  for (int ni = 0; ni < 4; ++ni)
    bp[ni] = w1t + (size_t)(64 * w + 16 * ni + lnm) * 1024 + hf * 16;

  const int srow = tid >> 2;        // staging: 0..63
  const int sc   = (tid & 3) * 8;   // staging: 0,8,16,24

  for (int k0 = 0; k0 < 1024; k0 += 32) {
    __syncthreads();
    // Stage 64x32 f32 chunk of concat(pair, rel) -> bf16 LDS tile.
    const float* src = (k0 < 512)
        ? (pairf + (r0 + srow) * 512 + (k0 + sc))
        : (relf  + (r0 + srow) * 512 + (k0 - 512 + sc));
    __builtin_prefetch(src + 32, 0, 1);   // next k-chunk (speculative, gfx1250 global_prefetch)
    const float4 f0 = *(const float4*)src;
    const float4 f1 = *(const float4*)(src + 4);
    BF16x8 st;
    st.p[0] = pk2(f0.x, f0.y); st.p[1] = pk2(f0.z, f0.w);
    st.p[2] = pk2(f1.x, f1.y); st.p[3] = pk2(f1.z, f1.w);
    *(v8bf*)&xtile[srow * XSTR + sc] = st.v;
    __syncthreads();

    // A fragments (contiguous K order): lanes<16 hold K 0-15;
    // lanes>=16 hold K 16-31.
    BF16x16 a[4];
#pragma unroll
    for (int mi = 0; mi < 4; ++mi) {
      const __bf16* p = &xtile[(16 * mi + lnm) * XSTR + hf * 16];
      a[mi].h[0] = *(const v8bf*)p;
      a[mi].h[1] = *(const v8bf*)(p + 8);
    }
#pragma unroll
    for (int ni = 0; ni < 4; ++ni) {
      BF16x16 b;   // lane holds col n, K = k0 + 16*hf + i (contiguous in n-major w1t)
      b.h[0] = *(const v8bf*)(bp[ni] + k0);
      b.h[1] = *(const v8bf*)(bp[ni] + k0 + 8);
#pragma unroll
      for (int mi = 0; mi < 4; ++mi)
        acc[mi * 4 + ni] = __builtin_amdgcn_wmma_f32_16x16x32_bf16(
            false, a[mi].v, false, b.v, (short)0, acc[mi * 4 + ni], false, false);
    }
  }
  __syncthreads();

  // bias + relu -> hbuf (bf16). C/D layout: element (M = v + 8*hf, N = lnm).
#pragma unroll
  for (int ni = 0; ni < 4; ++ni) {
    const int col = 64 * w + 16 * ni + lnm;
    const float bb = b1[col];
#pragma unroll
    for (int mi = 0; mi < 4; ++mi) {
      const int rbase = 16 * mi + 8 * hf;
#pragma unroll
      for (int v = 0; v < 8; v += 2) {
        float x0 = acc[mi * 4 + ni][v]     + bb; x0 = x0 > 0.f ? x0 : 0.f;
        float x1 = acc[mi * 4 + ni][v + 1] + bb; x1 = x1 > 0.f ? x1 : 0.f;
        const v2bf r = pk2(x0, x1);
        hbuf[(rbase + v) * HSTR + col]     = r[0];
        hbuf[(rbase + v + 1) * HSTR + col] = r[1];
      }
    }
  }
  __syncthreads();

  // GEMM2: 64x64 output = 16 tiles; each wave computes 2 tiles.
  const int ni2   = w & 3;
  const int mi2a  = (w >> 2) * 2;       // tiles mi2a, mi2a+1
  const int col2  = 16 * ni2 + lnm;
  const __bf16* bp2 = w2t + (size_t)col2 * 512 + hf * 16;
  v8f acc2[2] = {};
#pragma unroll
  for (int k0 = 0; k0 < 512; k0 += 32) {
    BF16x16 b;
    b.h[0] = *(const v8bf*)(bp2 + k0);
    b.h[1] = *(const v8bf*)(bp2 + k0 + 8);
#pragma unroll
    for (int t = 0; t < 2; ++t) {
      BF16x16 a;
      const __bf16* p = &hbuf[(16 * (mi2a + t) + lnm) * HSTR + k0 + hf * 16];
      a.h[0] = *(const v8bf*)p;
      a.h[1] = *(const v8bf*)(p + 8);
      acc2[t] = __builtin_amdgcn_wmma_f32_16x16x32_bf16(
          false, a.v, false, b.v, (short)0, acc2[t], false, false);
    }
  }
  if (col2 < RQ) {
    const float b2v = b2[col2];
#pragma unroll
    for (int t = 0; t < 2; ++t) {
#pragma unroll
      for (int v = 0; v < 8; ++v) {
        const float x = acc2[t][v] + b2v;
        const float s = 1.0f / (1.0f + __expf(-x));
        pred[(r0 + 16 * (mi2a + t) + v + 8 * hf) * RQ + col2] = s;
      }
    }
  }
}

// ---------------------------------------------------------------------------
// Kernel 3: focal-loss partial sums (fl1, fl0, num1) -> ws accumulators.
// ---------------------------------------------------------------------------
__global__ __launch_bounds__(256)
void th4_focal_loss(const float* __restrict__ pred, const int* __restrict__ gt,
                    float* __restrict__ accum) {
  float fl1 = 0.f, fl0 = 0.f, n1 = 0.f;
  for (size_t i = (size_t)blockIdx.x * blockDim.x + threadIdx.x; i < PRED_N;
       i += (size_t)gridDim.x * blockDim.x) {
    float p = pred[i];
    p = fminf(fmaxf(p, 1e-5f), 1.0f - 1e-5f);
    const float g = (float)gt[i];
    fl1 += (1.0f - p) * g * (-__logf(p));               // pos_gamma = 1
    fl0 += p * p * (1.0f - g) * (-__logf(1.0f - p));    // neg_gamma = 2
    n1  += g;
  }
  __shared__ float s[3];
  if (threadIdx.x < 3) s[threadIdx.x] = 0.f;
  __syncthreads();
  atomicAdd(&s[0], fl1); atomicAdd(&s[1], fl0); atomicAdd(&s[2], n1);
  __syncthreads();
  if (threadIdx.x == 0) {
    atomicAdd(&accum[0], s[0]); atomicAdd(&accum[1], s[1]); atomicAdd(&accum[2], s[2]);
  }
}

__global__ void th4_finalize_loss(const float* __restrict__ accum,
                                  float* __restrict__ loss_out) {
  if (threadIdx.x == 0 && blockIdx.x == 0) {
    const float fl1 = accum[0], fl0 = accum[1], n1 = accum[2];
    const float n0 = (float)PRED_N - n1;
    loss_out[0] = (n1 == 0.0f) ? fl1 : fl1 / fmaxf(n1, 1.0f);
    loss_out[1] = (n0 == 0.0f) ? fl0 : fl0 / fmaxf(n0, 1.0f);
  }
}

// ---------------------------------------------------------------------------
// Kernel 4: per-image metrics. Positive floats are uint-monotonic, so the
// exact k-th-largest value is found by 31-step binary search on bit patterns
// (count(x >= mid) is monotone). One block per image, 1024 threads.
// ---------------------------------------------------------------------------
__global__ __launch_bounds__(1024)
void th4_metrics(const float* __restrict__ pred, const int* __restrict__ gt,
                 float* __restrict__ out9) {
  const int img = blockIdx.x;
  const float* p = pred + (size_t)img * M_IMG;
  const int*   g = gt   + (size_t)img * M_IMG;
  const int tid = threadIdx.x;

  __shared__ unsigned scnt[4];
  __shared__ unsigned sg;
  if (tid == 0) sg = 0u;
  __syncthreads();
  unsigned gl = 0;
  for (int i = tid; i < M_IMG; i += 1024) gl += (unsigned)g[i];
  atomicAdd(&sg, gl);
  __syncthreads();
  const unsigned gtot = sg;

  unsigned ks[4];
  {
    int kg = (int)gtot; kg = kg < 1 ? 1 : (kg > M_IMG ? M_IMG : kg);
    ks[0] = (unsigned)kg; ks[1] = 20u; ks[2] = 50u; ks[3] = 100u;
  }
  unsigned lo[4], hi[4];
#pragma unroll
  for (int j = 0; j < 4; ++j) { lo[j] = 0u; hi[j] = 0x3F800001u; }  // (0, 1.0+ulp)

  for (int it = 0; it < 31; ++it) {
    unsigned mid[4];
#pragma unroll
    for (int j = 0; j < 4; ++j) mid[j] = lo[j] + ((hi[j] - lo[j]) >> 1);
    if (tid < 4) scnt[tid] = 0u;
    __syncthreads();
    unsigned c0 = 0, c1 = 0, c2 = 0, c3 = 0;
    for (int i = tid; i < M_IMG; i += 1024) {
      const unsigned b = __float_as_uint(p[i]);
      c0 += (b >= mid[0]); c1 += (b >= mid[1]); c2 += (b >= mid[2]); c3 += (b >= mid[3]);
    }
    atomicAdd(&scnt[0], c0); atomicAdd(&scnt[1], c1);
    atomicAdd(&scnt[2], c2); atomicAdd(&scnt[3], c3);
    __syncthreads();
#pragma unroll
    for (int j = 0; j < 4; ++j) { if (scnt[j] >= ks[j]) lo[j] = mid[j]; else hi[j] = mid[j]; }
    __syncthreads();
  }
  // lo[j] now holds bits of s[k_j - 1] (the >= threshold used by the reference).

  __shared__ unsigned stp[4], spc[4];
  if (tid < 4) { stp[tid] = 0u; spc[tid] = 0u; }
  __syncthreads();
  unsigned tp[4] = {0, 0, 0, 0}, pc[4] = {0, 0, 0, 0};
  for (int i = tid; i < M_IMG; i += 1024) {
    const unsigned b  = __float_as_uint(p[i]);
    const unsigned gi = (unsigned)g[i];
#pragma unroll
    for (int j = 0; j < 4; ++j) {
      const unsigned m = (b >= lo[j]) ? 1u : 0u;
      pc[j] += m; tp[j] += m * gi;
    }
  }
#pragma unroll
  for (int j = 0; j < 4; ++j) { atomicAdd(&stp[j], tp[j]); atomicAdd(&spc[j], pc[j]); }
  __syncthreads();
  if (tid == 0) {
    if (gtot > 0) {
#pragma unroll
      for (int j = 0; j < 4; ++j) {
        atomicAdd(&out9[j],     (float)stp[j]);   // tp, tp20, tp50, tp100
        atomicAdd(&out9[4 + j], (float)spc[j]);   // p, p20, p50, p100
      }
    }
    atomicAdd(&out9[8], (float)gtot);             // g
  }
}

// ---------------------------------------------------------------------------
extern "C" void kernel_launch(void* const* d_in, const int* in_sizes, int n_in,
                              void* d_out, int out_size, void* d_ws, size_t ws_size,
                              hipStream_t stream) {
  const float* pairf = (const float*)d_in[0];
  const float* relf  = (const float*)d_in[1];
  const int*   gt    = (const int*)d_in[2];
  const float* w1    = (const float*)d_in[3];
  const float* b1    = (const float*)d_in[4];
  const float* w2    = (const float*)d_in[5];
  const float* b2    = (const float*)d_in[6];
  float* out = (float*)d_out;

  char* ws = (char*)d_ws;
  __bf16* w1t  = (__bf16*)ws;                       // 512*1024*2 = 1 MB
  __bf16* w2t  = (__bf16*)(ws + 1048576);           // 64*512*2 = 64 KB
  float*  wsac = (float*)(ws + 1048576 + 65536);    // 3 f32 accumulators

  float* pred        = out;
  float* loss_out    = out + PRED_N;                // 2 floats
  float* metrics_out = out + PRED_N + 2;            // 9 floats

  {
    const int total  = 512 * 1024 + 64 * 512 + 11 + 3;
    const int blocks = (total + 255) / 256;
    th4_convert<<<blocks, 256, 0, stream>>>(w1, w2, w1t, w2t, wsac, loss_out);
  }
  th4_gemm_fused<<<M_TOTAL / MTILE, 256, LDS_TOTAL_B, stream>>>(
      pairf, relf, w1t, b1, w2t, b2, pred);
  th4_focal_loss<<<2048, 256, 0, stream>>>(pred, gt, wsac);
  th4_finalize_loss<<<1, 1, 0, stream>>>(wsac, loss_out);
  th4_metrics<<<BQ, 1024, 0, stream>>>(pred, gt, metrics_out);
}
